// ResonanceTransformer_5308579578078
// MI455X (gfx1250) — matrix operand order repaired and checked by reading, not compiled, not run on hardware
//
#include <hip/hip_runtime.h>
#include <hip/hip_bf16.h>

// ---------------------------------------------------------------------------
// ResonanceTransformer forward for MI455X (gfx1250, wave32, WMMA).
// GEMMs: v_wmma_f32_16x16x32_f16, LDS panels double-buffered via
// global_load_async_to_lds_b128 + s_wait_asynccnt (CDNA5 async path).
// ---------------------------------------------------------------------------

#define PI_F        3.14159265358979323846f
#define THIRD_F     2.09439510239319549231f   // 2*pi/3
#define PI2_F       9.86960440108935861883f   // pi^2
#define DEV_OVER_PI 0.00318309886183790672f   // 0.01/pi
#define SQRT_D      22.6274169979695223f      // sqrt(512)

#define BB   16
#define SS   512
#define DD   512
#define HH   8
#define DHH  64
#define FF   2048
#define LL   6
#define VV   10000
#define VPAD 10112            // 79*128
#define MR   (BB*SS)          // 8192 rows of activations

typedef __attribute__((ext_vector_type(16))) _Float16 v16h;
typedef __attribute__((ext_vector_type(8)))  float    v8f;

static __device__ __forceinline__ v8f wmma16(v16h a, v16h b, v8f c) {
  return __builtin_amdgcn_wmma_f32_16x16x32_f16(false, a, false, b, (short)0, c,
                                                false, false);
}

// --------------------------- embedding -------------------------------------
__global__ __launch_bounds__(256)
void k_embed(const int* __restrict__ src, const float* __restrict__ emb,
             const float* __restrict__ pos, float* __restrict__ x,
             _Float16* __restrict__ xh) {
  int row = blockIdx.x;                 // 0..8191
  int t   = threadIdx.x;
  int tok = src[row];
  int s   = row & (SS - 1);
#pragma unroll
  for (int j = 0; j < 2; ++j) {
    int d = t + j * 256;
    float v = emb[(size_t)tok * DD + d] * SQRT_D + pos[(size_t)s * DD + d];
    x[(size_t)row * DD + d]  = v;
    xh[(size_t)row * DD + d] = (_Float16)v;
  }
}

// --------------------------- prune: resonance chain ------------------------
__global__ __launch_bounds__(256)
void k_etch(const float* __restrict__ w, float* __restrict__ out, int n,
            float inv_sm1) {
  int i = blockIdx.x * 256 + threadIdx.x;
  if (i >= n) return;
  float ww    = w[i];
  float bloom = __sinf(ww * PI_F);                       // already in [-1,1]
  float ang   = (float)i * inv_sm1 + THIRD_F;            // theta/s + THIRD
  bloom       = bloom + bloom * __cosf(ang) * 1.5f;
  float e     = __cosf(bloom * PI2_F) + bloom * bloom * DEV_OVER_PI;
  out[i]      = fabsf(e);
}

// --------------------------- exact quantile: radix select ------------------
// sb layout (uint32): [0..255] hist, [256] prefix, [257] kleft,
//                     [258] bits(k0), [259] bits(k1), [260] thr (as float)
__global__ void k_sel_init(unsigned* sb, unsigned k) {
  int t = threadIdx.x;
  if (t < 256) sb[t] = 0u;
  if (t == 0) { sb[256] = 0u; sb[257] = k; }
}

__global__ __launch_bounds__(256)
void k_sel_hist(const float* __restrict__ v, int n, unsigned* __restrict__ sb,
                int shift) {
  __shared__ unsigned lh[256];
  int t = threadIdx.x;
  lh[t] = 0u;
  __syncthreads();
  unsigned pfx = sb[256];
  for (int i = blockIdx.x * 256 + t; i < n; i += gridDim.x * 256) {
    unsigned u = __float_as_uint(v[i]);
    bool cand = (shift == 24) || ((u >> (shift + 8)) == (pfx >> (shift + 8)));
    if (cand) atomicAdd(&lh[(u >> shift) & 255u], 1u);
  }
  __syncthreads();
  if (lh[t]) atomicAdd(&sb[t], lh[t]);
}

__global__ void k_sel_scan(unsigned* sb, int shift, int outIdx) {
  if (threadIdx.x != 0 || blockIdx.x != 0) return;
  unsigned k = sb[257], cum = 0, d = 0;
  for (d = 0; d < 256; ++d) {
    unsigned c = sb[d];
    if (k < cum + c) break;
    cum += c;
  }
  if (d > 255) d = 255;
  sb[256] |= d << shift;
  sb[257]  = k - cum;
  for (int i = 0; i < 256; ++i) sb[i] = 0u;
  if (shift == 0) sb[outIdx] = sb[256];
}

__global__ void k_sel_thr(unsigned* sb) {
  if (threadIdx.x != 0 || blockIdx.x != 0) return;
  float lo = __uint_as_float(sb[258]);
  float hi = __uint_as_float(sb[259]);
  ((float*)sb)[260] = lo + 0.75f * (hi - lo);   // linear interp, frac = 0.75
}

// mask (strict >) fused with f32 -> f16 conversion
__global__ __launch_bounds__(256)
void k_maskcvt(const float* __restrict__ w, const float* __restrict__ etch,
               const unsigned* __restrict__ sb, _Float16* __restrict__ wh,
               int n) {
  int i = blockIdx.x * 256 + threadIdx.x;
  if (i >= n) return;
  float thr = ((const float*)sb)[260];
  wh[i] = (etch[i] > thr) ? (_Float16)w[i] : (_Float16)0.0f;
}

// plain f32 -> f16 conversion with row zero-padding (for unpruned weights)
__global__ __launch_bounds__(256)
void k_cvtpad(const float* __restrict__ w, _Float16* __restrict__ wh,
              int Nreal, int Npad, int K) {
  int i = blockIdx.x * 256 + threadIdx.x;
  if (i >= Npad * K) return;
  int row = i / K;
  wh[i] = (row < Nreal) ? (_Float16)w[i] : (_Float16)0.0f;
}

// --------------------------- WMMA tiled GEMM -------------------------------
// C[M,N] = A[M,K] @ W[N,K]^T + bias[N]
// MODE 0: f16 out; MODE 1: f16 relu out; MODE 2: f32 out
// LDS panels are double-buffered and filled with async direct-to-LDS copies.
#define PANEL (128 * 40)      // halves per (A or W) panel, row-padded to 40

template <int MODE>
__global__ __launch_bounds__(256)
void k_gemm(const _Float16* __restrict__ A, const _Float16* __restrict__ W,
            const float* __restrict__ bias, float* __restrict__ Cf,
            _Float16* __restrict__ Ch, int N, int K, int ldc) {
  __shared__ __align__(16) _Float16 smem[2 * 2 * PANEL];  // [buf][A|W][128][40]
  const int t  = threadIdx.x;
  const int wv = t >> 5, ln = t & 31;
  const int nn = ln & 15, hf = ln >> 4;
  const int m0 = blockIdx.y * 128, n0 = blockIdx.x * 128;

  v8f acc[8] = {};

  const int r  = t >> 1;                  // 0..127: panel row this thread fills
  const int ch = (t & 1) << 4;            // 0 or 16 halves within the row

  // Issue 4 async 16B copies (2 for A, 2 for W) into LDS buffer `b`,
  // covering this thread's 16 halves of each 128x32 panel for K-step `kb`.
  auto issue = [&](int b, int kb) {
    const _Float16* gA = A + (size_t)(m0 + r) * K + kb * 32 + ch;
    const _Float16* gW = W + (size_t)(n0 + r) * K + kb * 32 + ch;
    _Float16* lA = smem + b * (2 * PANEL) + r * 40 + ch;
    _Float16* lW = smem + b * (2 * PANEL) + PANEL + r * 40 + ch;
    unsigned la = (unsigned)(unsigned long long)lA;  // LDS byte offset
    unsigned lw = (unsigned)(unsigned long long)lW;
    asm volatile(
        "global_load_async_to_lds_b128 %0, %1, off\n\t"
        "global_load_async_to_lds_b128 %0, %1, off offset:16"
        :: "v"(la), "v"(gA) : "memory");
    asm volatile(
        "global_load_async_to_lds_b128 %0, %1, off\n\t"
        "global_load_async_to_lds_b128 %0, %1, off offset:16"
        :: "v"(lw), "v"(gW) : "memory");
  };

  const int nk = K >> 5;
  issue(0, 0);
  for (int kb = 0; kb < nk; ++kb) {
    const int cur = kb & 1;
    if (kb + 1 < nk) {
      issue(cur ^ 1, kb + 1);
      // 4 new copies in flight; wait until only those remain -> panel `cur`
      // is fully resident (async loads complete in order).
      asm volatile("s_wait_asynccnt 0x4" ::: "memory");
    } else {
      asm volatile("s_wait_asynccnt 0x0" ::: "memory");
    }
    __syncthreads();                       // all waves' copies visible

    const _Float16* As = smem + cur * (2 * PANEL);
    const _Float16* Ws = As + PANEL;

    v16h af;
#pragma unroll
    for (int i = 0; i < 16; ++i)
      af[i] = As[(wv * 16 + nn) * 40 + ((i >> 3) << 4) + hf * 8 + (i & 7)];
#pragma unroll
    for (int j = 0; j < 8; ++j) {
      v16h bf;
#pragma unroll
      for (int i = 0; i < 16; ++i)
        bf[i] = Ws[(j * 16 + nn) * 40 + ((i >> 3) << 4) + hf * 8 + (i & 7)];
      acc[j] = wmma16(af, bf, acc[j]);
    }
    __syncthreads();   // everyone done reading `cur` before it is refilled
  }

  const int row0 = m0 + wv * 16 + hf * 8;
#pragma unroll
  for (int j = 0; j < 8; ++j) {
    int col = n0 + j * 16 + nn;
    bool ok = col < N;
    float bv = ok ? bias[col] : 0.0f;
#pragma unroll
    for (int rr = 0; rr < 8; ++rr) {
      float v = acc[j][rr] + bv;
      if (MODE == 1) v = v > 0.0f ? v : 0.0f;
      if (ok) {
        size_t idx = (size_t)(row0 + rr) * ldc + col;
        if (MODE == 2) Cf[idx] = v;
        else           Ch[idx] = (_Float16)v;
      }
    }
  }
}

// --------------------------- flash attention (1 wave / 16 queries) ---------
__global__ __launch_bounds__(32)
void k_attn(const _Float16* __restrict__ qkv, _Float16* __restrict__ oh) {
  __shared__ _Float16 Pl[16 * 32];
  const int ln = threadIdx.x, nn = ln & 15, hf = ln >> 4;
  const int q0 = blockIdx.x * 16, h = blockIdx.y, b = blockIdx.z;
  const size_t rs = 3 * DD;                            // qkv row stride
  const _Float16* Qp = qkv + (size_t)b * SS * rs + h * DHH;
  const _Float16* Kp = Qp + DD;
  const _Float16* Vp = Qp + 2 * DD;

  v16h aq0, aq1;
#pragma unroll
  for (int i = 0; i < 16; ++i) {
    int k = ((i >> 3) << 4) + hf * 8 + (i & 7);
    aq0[i] = Qp[(size_t)(q0 + nn) * rs + k];
    aq1[i] = Qp[(size_t)(q0 + nn) * rs + 32 + k];
  }

  float mrun[8], lrun[8];
  v8f o[4] = {};
#pragma unroll
  for (int rr = 0; rr < 8; ++rr) { mrun[rr] = -1e30f; lrun[rr] = 0.0f; }

  for (int kb = 0; kb < SS; kb += 32) {
    v16h bk00, bk10, bk01, bk11;
#pragma unroll
    for (int i = 0; i < 16; ++i) {
      int c = ((i >> 3) << 4) + hf * 8 + (i & 7);
      bk00[i] = Kp[(size_t)(kb + nn) * rs + c];
      bk10[i] = Kp[(size_t)(kb + nn) * rs + 32 + c];
      bk01[i] = Kp[(size_t)(kb + 16 + nn) * rs + c];
      bk11[i] = Kp[(size_t)(kb + 16 + nn) * rs + 32 + c];
    }
    v8f s0 = {}, s1 = {};
    s0 = wmma16(aq0, bk00, s0);
    s0 = wmma16(aq1, bk10, s0);
    s1 = wmma16(aq0, bk01, s1);
    s1 = wmma16(aq1, bk11, s1);

#pragma unroll
    for (int rr = 0; rr < 8; ++rr) {
      float a = s0[rr] * 0.125f;                       // 1/sqrt(64)
      float c = s1[rr] * 0.125f;
      float red = fmaxf(a, c);
#pragma unroll
      for (int off = 1; off < 16; off <<= 1)
        red = fmaxf(red, __shfl_xor(red, off, 32));
      float mnew  = fmaxf(mrun[rr], red);
      float alpha = __expf(mrun[rr] - mnew);
      float p0 = __expf(a - mnew), p1 = __expf(c - mnew);
      float rsum = p0 + p1;
#pragma unroll
      for (int off = 1; off < 16; off <<= 1)
        rsum += __shfl_xor(rsum, off, 32);
      lrun[rr] = lrun[rr] * alpha + rsum;
      mrun[rr] = mnew;
#pragma unroll
      for (int j = 0; j < 4; ++j) o[j][rr] *= alpha;
      Pl[(hf * 8 + rr) * 32 + nn]      = (_Float16)p0;
      Pl[(hf * 8 + rr) * 32 + 16 + nn] = (_Float16)p1;
    }
    __syncthreads();

    v16h pa;
#pragma unroll
    for (int i = 0; i < 16; ++i)
      pa[i] = Pl[nn * 32 + ((i >> 3) << 4) + hf * 8 + (i & 7)];
#pragma unroll
    for (int j = 0; j < 4; ++j) {
      v16h bv;
#pragma unroll
      for (int i = 0; i < 16; ++i)
        bv[i] = Vp[(size_t)(kb + ((i >> 3) << 4) + hf * 8 + (i & 7)) * rs + j * 16 + nn];
      o[j] = wmma16(pa, bv, o[j]);
    }
    __syncthreads();
  }

#pragma unroll
  for (int rr = 0; rr < 8; ++rr) {
    float inv = 1.0f / lrun[rr];
    int row = q0 + hf * 8 + rr;
#pragma unroll
    for (int j = 0; j < 4; ++j)
      oh[(size_t)(b * SS + row) * DD + h * DHH + j * 16 + nn] =
          (_Float16)(o[j][rr] * inv);
  }
}

// --------------------------- residual + layernorm --------------------------
__global__ __launch_bounds__(256)
void k_add_ln(float* __restrict__ x, const float* __restrict__ add,
              const float* __restrict__ g, const float* __restrict__ bta,
              _Float16* __restrict__ xh) {
  __shared__ float red[256];
  int row = blockIdx.x, t = threadIdx.x;
  float v0 = x[(size_t)row * DD + t]       + add[(size_t)row * DD + t];
  float v1 = x[(size_t)row * DD + 256 + t] + add[(size_t)row * DD + 256 + t];

  red[t] = v0 + v1;
  __syncthreads();
  for (int o2 = 128; o2 > 0; o2 >>= 1) {
    if (t < o2) red[t] += red[t + o2];
    __syncthreads();
  }
  float mean = red[0] * (1.0f / DD);
  __syncthreads();
  red[t] = v0 * v0 + v1 * v1;
  __syncthreads();
  for (int o2 = 128; o2 > 0; o2 >>= 1) {
    if (t < o2) red[t] += red[t + o2];
    __syncthreads();
  }
  float var  = red[0] * (1.0f / DD) - mean * mean;
  float rstd = rsqrtf(var + 1e-5f);
  float y0 = (v0 - mean) * rstd * g[t]       + bta[t];
  float y1 = (v1 - mean) * rstd * g[256 + t] + bta[256 + t];
  x[(size_t)row * DD + t]        = y0;
  x[(size_t)row * DD + 256 + t]  = y1;
  xh[(size_t)row * DD + t]       = (_Float16)y0;
  xh[(size_t)row * DD + 256 + t] = (_Float16)y1;
}

// --------------------------- host orchestration ----------------------------
static void run_select(const float* v, int n, unsigned k, unsigned* sb,
                       int outIdx, hipStream_t s) {
  k_sel_init<<<1, 256, 0, s>>>(sb, k);
  int blocks = (n + 255) / 256;
  if (blocks > 4096) blocks = 4096;
  for (int shift = 24; shift >= 0; shift -= 8) {
    k_sel_hist<<<blocks, 256, 0, s>>>(v, n, sb, shift);
    k_sel_scan<<<1, 1, 0, s>>>(sb, shift, outIdx);
  }
}

static void run_prune(const float* w, int n, float* etch, unsigned* sb,
                      _Float16* wh, hipStream_t s) {
  int blocks = (n + 255) / 256;
  k_etch<<<blocks, 256, 0, s>>>(w, etch, n, 1.0f / (float)(n - 1));
  run_select(etch, n, (unsigned)(n / 4 - 1), sb, 258, s);
  run_select(etch, n, (unsigned)(n / 4),     sb, 259, s);
  k_sel_thr<<<1, 1, 0, s>>>(sb);
  k_maskcvt<<<blocks, 256, 0, s>>>(w, etch, sb, wh, n);
}

extern "C" void kernel_launch(void* const* d_in, const int* in_sizes, int n_in,
                              void* d_out, int out_size, void* d_ws,
                              size_t ws_size, hipStream_t stream) {
  (void)in_sizes; (void)n_in; (void)out_size; (void)ws_size;

  const int*   src        = (const int*)  d_in[0];
  const float* emb        = (const float*)d_in[1];
  const float* pos        = (const float*)d_in[2];
  const float* in_proj_w  = (const float*)d_in[3];
  const float* in_proj_b  = (const float*)d_in[4];
  const float* out_proj_w = (const float*)d_in[5];
  const float* out_proj_b = (const float*)d_in[6];
  const float* ln1_g      = (const float*)d_in[7];
  const float* ln1_b      = (const float*)d_in[8];
  const float* ln2_g      = (const float*)d_in[9];
  const float* ln2_b      = (const float*)d_in[10];
  const float* lin1_w     = (const float*)d_in[11];
  const float* lin1_b     = (const float*)d_in[12];
  const float* lin2_w     = (const float*)d_in[13];
  const float* lin2_b     = (const float*)d_in[14];
  const float* out_w      = (const float*)d_in[15];
  const float* out_b      = (const float*)d_in[16];
  float* logits = (float*)d_out;

  // ---- carve workspace ----
  char* p = (char*)d_ws;
  auto carve = [&](size_t bytes) {
    void* q = p;
    p += (bytes + 255) & ~(size_t)255;
    return q;
  };
  float*    x     = (float*)   carve((size_t)MR * DD * 4);     // 16 MB
  _Float16* xh    = (_Float16*)carve((size_t)MR * DD * 2);     //  8 MB
  _Float16* qkvh  = (_Float16*)carve((size_t)MR * 3 * DD * 2); // 24 MB
  _Float16* oh    = (_Float16*)carve((size_t)MR * DD * 2);     //  8 MB
  _Float16* ff1h  = (_Float16*)carve((size_t)MR * FF * 2);     // 32 MB
  float*    tmpf  = (float*)   carve((size_t)MR * DD * 4);     // 16 MB
  float*    etch  = (float*)   carve((size_t)FF * DD * 4);     //  4 MB
  _Float16* wh    = (_Float16*)carve((size_t)VPAD * DD * 2);   // 10.4 MB
  unsigned* sb    = (unsigned*)carve(4096);

  // ---- embedding ----
  k_embed<<<MR, 256, 0, stream>>>(src, emb, pos, x, xh);

  // ---- layers ----
  for (int l = 0; l < LL; ++l) {
    // QKV projection (pruned weight)
    run_prune(in_proj_w + (size_t)l * 3 * DD * DD, 3 * DD * DD, etch, sb, wh,
              stream);
    k_gemm<0><<<dim3(12, 64), 256, 0, stream>>>(
        xh, wh, in_proj_b + (size_t)l * 3 * DD, nullptr, qkvh,
        3 * DD, DD, 3 * DD);

    // attention
    k_attn<<<dim3(SS / 16, HH, BB), 32, 0, stream>>>(qkvh, oh);

    // output projection (not pruned)
    k_cvtpad<<<(DD * DD + 255) / 256, 256, 0, stream>>>(
        out_proj_w + (size_t)l * DD * DD, wh, DD, DD, DD);
    k_gemm<2><<<dim3(4, 64), 256, 0, stream>>>(
        oh, wh, out_proj_b + (size_t)l * DD, tmpf, nullptr, DD, DD, DD);

    k_add_ln<<<MR, 256, 0, stream>>>(x, tmpf, ln1_g + (size_t)l * DD,
                                     ln1_b + (size_t)l * DD, xh);

    // FFN up (pruned, relu)
    run_prune(lin1_w + (size_t)l * FF * DD, FF * DD, etch, sb, wh, stream);
    k_gemm<1><<<dim3(16, 64), 256, 0, stream>>>(
        xh, wh, lin1_b + (size_t)l * FF, nullptr, ff1h, FF, DD, FF);

    // FFN down (pruned)
    run_prune(lin2_w + (size_t)l * DD * FF, DD * FF, etch, sb, wh, stream);
    k_gemm<2><<<dim3(4, 64), 256, 0, stream>>>(
        ff1h, wh, lin2_b + (size_t)l * DD, tmpf, nullptr, DD, FF, DD);

    k_add_ln<<<MR, 256, 0, stream>>>(x, tmpf, ln2_g + (size_t)l * DD,
                                     ln2_b + (size_t)l * DD, xh);
  }

  // ---- output head (padded to 10112 rows for 128-aligned tiling) ----
  k_cvtpad<<<((size_t)VPAD * DD + 255) / 256, 256, 0, stream>>>(
      out_w, wh, VV, VPAD, DD);
  k_gemm<2><<<dim3(VPAD / 128, 64), 256, 0, stream>>>(
      xh, wh, out_b, logits, nullptr, VV, DD, VV);
}
